// MultiHeadAttention_90975997264381
// MI455X (gfx1250) — matrix operand order repaired and checked by reading, not compiled, not run on hardware
//
#include <hip/hip_runtime.h>
#include <cmath>

#define B_DIM 2
#define S_LEN 2048
#define E_DIM 1024
#define NH    16
#define HD    64
#define M_DIM (B_DIM * S_LEN)   // 4096 rows

typedef __attribute__((ext_vector_type(16))) _Float16 v16h;
typedef __attribute__((ext_vector_type(8)))  _Float16 v8h;
typedef __attribute__((ext_vector_type(8)))  float    v8f;

// ---------------------------------------------------------------------------
// WMMA fragment loaders (CDNA5 ISA 7.12.2 layouts, wave32)
// ---------------------------------------------------------------------------
__device__ __forceinline__ v16h load_a_frag(const _Float16* base, int stride, int k0) {
  const int l   = threadIdx.x & 31;
  const int row = l & 15;
  const int hi  = l >> 4;
  const _Float16* p = base + row * stride + k0 + hi * 8;
  v8h lo = *(const v8h*)(p);
  v8h hc = *(const v8h*)(p + 16);
  return __builtin_shufflevector(lo, hc, 0,1,2,3,4,5,6,7,8,9,10,11,12,13,14,15);
}

__device__ __forceinline__ v16h load_b_frag(const _Float16* base, int stride, int k0) {
  const int l   = threadIdx.x & 31;
  const int col = l & 15;
  const int hi  = l >> 4;
  const _Float16* p = base + col * stride + k0 + hi * 8;
  v8h lo = *(const v8h*)(p);
  v8h hc = *(const v8h*)(p + 16);
  return __builtin_shufflevector(lo, hc, 0,1,2,3,4,5,6,7,8,9,10,11,12,13,14,15);
}

__device__ __forceinline__ v8f wmma_f16(v16h a, v16h b, v8f c) {
  return __builtin_amdgcn_wmma_f32_16x16x32_f16(false, a, false, b, (short)0, c,
                                                false, false);
}

// ---------------------------------------------------------------------------
// CDNA5 async global->LDS copies (ASYNCcnt-tracked, bypass VGPRs).
// ---------------------------------------------------------------------------
__device__ __forceinline__ void async_copy_b128(_Float16* lds_dst,
                                                const _Float16* gsrc) {
  const unsigned laddr = (unsigned)(uintptr_t)lds_dst;
  asm volatile("global_load_async_to_lds_b128 %0, %1, off"
               :: "v"(laddr), "v"(gsrc) : "memory");
}
__device__ __forceinline__ void async_wait_all() {
  asm volatile("s_wait_asynccnt 0" ::: "memory");
}
__device__ __forceinline__ void async_wait_1() {
  asm volatile("s_wait_asynccnt 1" ::: "memory");
}
__device__ __forceinline__ void async_wait_2() {
  asm volatile("s_wait_asynccnt 2" ::: "memory");
}
__device__ __forceinline__ void async_wait_8() {
  asm volatile("s_wait_asynccnt 8" ::: "memory");
}

// ---------------------------------------------------------------------------
// Kernel 0: one-time weight convert+transpose: Wt[n][k] = (f16) W[k][n]
// ---------------------------------------------------------------------------
__global__ void __launch_bounds__(256)
wconvert_kernel(const float* __restrict__ W, _Float16* __restrict__ Wt)
{
  __shared__ float T[32][33];
  const int k0 = blockIdx.x * 32;
  const int n0 = blockIdx.y * 32;
  const int tid = threadIdx.x;
  const int c  = tid & 31;
  const int r4 = tid >> 5;           // 0..7
  #pragma unroll
  for (int rr = 0; rr < 32; rr += 8)
    T[rr + r4][c] = W[(size_t)(k0 + rr + r4) * E_DIM + n0 + c];
  __syncthreads();
  #pragma unroll
  for (int rr = 0; rr < 32; rr += 8)
    Wt[(size_t)(n0 + rr + r4) * E_DIM + k0 + c] = (_Float16)T[c][rr + r4];
}

// ---------------------------------------------------------------------------
// Kernel 1: X[M,E](f32) @ W + bias -> f16 head layout.
// B tiles (pre-transposed f16 weights) stream in via async double-buffered
// LDS copies; A tiles pipeline through registers (cvt after the WMMAs).
// STORE_T == 0 : out[b][h][s][d]   (Q, K)
// STORE_T == 1 : out[b][h][d][s]   (V transposed for async attention loads)
// ---------------------------------------------------------------------------
template <int STORE_T>
__global__ void __launch_bounds__(256)
qkv_proj_kernel(const float* __restrict__ X, const _Float16* __restrict__ Wt,
                const float* __restrict__ bias, _Float16* __restrict__ outh,
                float scale)
{
  __shared__ _Float16 As[2][64 * 40];
  __shared__ _Float16 Bs[2][64 * 40];

  const int m0  = blockIdx.x * 64;
  const int n0  = blockIdx.y * 64;
  const int tid = threadIdx.x;
  const int wave = tid >> 5;
  const int tr   = wave >> 1;
  const int tc0  = (wave & 1) * 2;

  v8f acc0 = {}; v8f acc1 = {};

  const int ar = tid >> 3;             // 0..31 (A stage rows, +32 second half)
  const int ac = (tid & 7) * 4;        // 0..28
  const int br = tid >> 2;             // 0..63 (B async chunk row)
  const int bc = (tid & 3) * 8;        // 0..24

  // ---- prologue: stage 0 ----
  async_copy_b128(Bs[0] + br * 40 + bc, Wt + (size_t)(n0 + br) * E_DIM + bc);
  {
    float4 f0 = *(const float4*)(X + (size_t)(m0 + ar) * E_DIM + ac);
    float4 f1 = *(const float4*)(X + (size_t)(m0 + ar + 32) * E_DIM + ac);
    _Float16* d0 = As[0] + ar * 40 + ac;
    d0[0]=(_Float16)f0.x; d0[1]=(_Float16)f0.y; d0[2]=(_Float16)f0.z; d0[3]=(_Float16)f0.w;
    _Float16* d1 = As[0] + (ar + 32) * 40 + ac;
    d1[0]=(_Float16)f1.x; d1[1]=(_Float16)f1.y; d1[2]=(_Float16)f1.z; d1[3]=(_Float16)f1.w;
  }

  const int NK = E_DIM / 32;
  for (int it = 0; it < NK; ++it) {
    const int cur = it & 1, nxt = cur ^ 1;
    const bool pre = (it + 1) < NK;
    float4 fa0, fa1;
    if (pre) {
      const int k1 = (it + 1) * 32;
      async_copy_b128(Bs[nxt] + br * 40 + bc,
                      Wt + (size_t)(n0 + br) * E_DIM + k1 + bc);
      fa0 = *(const float4*)(X + (size_t)(m0 + ar) * E_DIM + k1 + ac);
      fa1 = *(const float4*)(X + (size_t)(m0 + ar + 32) * E_DIM + k1 + ac);
      async_wait_1();
    } else {
      async_wait_all();
    }
    __syncthreads();                       // stage `it` published

    v16h a  = load_a_frag(As[cur] + tr * 16 * 40, 40, 0);
    v16h b0 = load_b_frag(Bs[cur] + tc0 * 16 * 40, 40, 0);
    v16h b1 = load_b_frag(Bs[cur] + (tc0 + 1) * 16 * 40, 40, 0);
    acc0 = wmma_f16(a, b0, acc0);
    acc1 = wmma_f16(a, b1, acc1);

    if (pre) {                             // cvt after compute: loadcnt wait lands here
      _Float16* d0 = As[nxt] + ar * 40 + ac;
      d0[0]=(_Float16)fa0.x; d0[1]=(_Float16)fa0.y; d0[2]=(_Float16)fa0.z; d0[3]=(_Float16)fa0.w;
      _Float16* d1 = As[nxt] + (ar + 32) * 40 + ac;
      d1[0]=(_Float16)fa1.x; d1[1]=(_Float16)fa1.y; d1[2]=(_Float16)fa1.z; d1[3]=(_Float16)fa1.w;
    }
    __syncthreads();                       // all readers done before refill
  }

  const int l  = tid & 31;
  const int hi = l >> 4;
  const int nn = l & 15;
  #pragma unroll
  for (int t = 0; t < 2; ++t) {
    const v8f acc = t ? acc1 : acc0;
    const int gn = n0 + (tc0 + t) * 16 + nn;
    const float bv = bias[gn];
    const int h = gn >> 6, d = gn & (HD - 1);
    #pragma unroll
    for (int r = 0; r < 8; ++r) {
      const int gm = m0 + tr * 16 + hi * 8 + r;
      const int bb = gm >> 11, s = gm & (S_LEN - 1);
      const _Float16 val = (_Float16)((acc[r] + bv) * scale);
      if (STORE_T)
        outh[(((size_t)bb * NH + h) * HD + d) * S_LEN + s] = val;
      else
        outh[(((size_t)bb * NH + h) * S_LEN + s) * HD + d] = val;
    }
  }
}

// ---------------------------------------------------------------------------
// Kernel 2: flash attention, 128 threads = 4 waves per 64 query rows of one
// (b,h). K/V tiles double-buffered via async global->LDS copies overlapping
// the WMMA stream. V comes pre-transposed from global ([b,h,d,s]).
// ---------------------------------------------------------------------------
__global__ void __launch_bounds__(128)
attn_kernel(const _Float16* __restrict__ Qh, const _Float16* __restrict__ Kh,
            const _Float16* __restrict__ Vt, _Float16* __restrict__ Ctx,
            const int* __restrict__ maskp)
{
  __shared__ _Float16 Qs [64 * 72];        // [qrow][d]
  __shared__ _Float16 Ks [2][64 * 72];     // [key][d]
  __shared__ _Float16 Vts[2][64 * 72];     // [d][key]
  __shared__ _Float16 Ps [4][16 * 72];     // per-wave P strip [qrow][key]

  const int causal = maskp[0];
  const int qb = blockIdx.x;
  const int bh = blockIdx.y;
  const int b  = bh >> 4;
  const int h  = bh & (NH - 1);
  const int tid  = threadIdx.x;
  const int wave = tid >> 5;
  const int l  = tid & 31;
  const int hi = l >> 4;
  const int nn = l & 15;

  const _Float16* Qg  = Qh + ((size_t)bh * S_LEN + qb * 64) * HD;
  const _Float16* Kg0 = Kh + (size_t)bh * S_LEN * HD;
  const _Float16* Vg0 = Vt + (size_t)bh * HD * S_LEN;   // rows are d, cols s

  #pragma unroll
  for (int i = 0; i < 4; ++i) {
    int chunk = i * 128 + tid;
    int r = chunk >> 3, c = (chunk & 7) * 8;
    async_copy_b128(Qs + r * 72 + c, Qg + r * 64 + c);
  }

  const int kb_max = causal ? qb : (S_LEN / 64 - 1);

  #pragma unroll
  for (int i = 0; i < 4; ++i) {
    int chunk = i * 128 + tid;
    int r = chunk >> 3, c = (chunk & 7) * 8;
    async_copy_b128(Ks[0]  + r * 72 + c, Kg0 + (size_t)r * HD + c);
    async_copy_b128(Vts[0] + r * 72 + c, Vg0 + (size_t)r * S_LEN + c);
  }

  v8f Oacc[4] = {};
  float m_i[8], l_i[8];
  #pragma unroll
  for (int r = 0; r < 8; ++r) { m_i[r] = -1e30f; l_i[r] = 0.0f; }

  for (int kb = 0; kb <= kb_max; ++kb) {
    const int cur = kb & 1;
    if (kb < kb_max) {
      const int nxt = (kb + 1) & 1;
      const _Float16* Kg = Kg0 + (size_t)(kb + 1) * 64 * HD;
      const _Float16* Vg = Vg0 + (size_t)(kb + 1) * 64;
      #pragma unroll
      for (int i = 0; i < 4; ++i) {
        int chunk = i * 128 + tid;
        int r = chunk >> 3, c = (chunk & 7) * 8;
        async_copy_b128(Ks[nxt]  + r * 72 + c, Kg + (size_t)r * HD + c);
        async_copy_b128(Vts[nxt] + r * 72 + c, Vg + (size_t)r * S_LEN + c);
      }
      async_wait_8();                      // stage kb (and Q) landed
    } else {
      async_wait_all();
    }
    __syncthreads();

    // S = Q . K^T  (inner dim d = 64 -> two x32 WMMA steps)
    v16h qa0 = load_a_frag(Qs + wave * 16 * 72, 72, 0);
    v16h qa1 = load_a_frag(Qs + wave * 16 * 72, 72, 32);
    v8f sc[4];
    #pragma unroll
    for (int tc = 0; tc < 4; ++tc) {
      v8f sv = {};
      sv = wmma_f16(qa0, load_b_frag(Ks[cur] + tc * 16 * 72, 72, 0),  sv);
      sv = wmma_f16(qa1, load_b_frag(Ks[cur] + tc * 16 * 72, 72, 32), sv);
      sc[tc] = sv;
    }

    const int qrow0 = qb * 64 + wave * 16 + hi * 8;
    if (causal) {
      #pragma unroll
      for (int tc = 0; tc < 4; ++tc) {
        const int kg = kb * 64 + tc * 16 + nn;
        #pragma unroll
        for (int r = 0; r < 8; ++r)
          if (kg > qrow0 + r) sc[tc][r] = -1e30f;
      }
    }

    // online softmax: rows live in 16-lane half-groups -> shfl_xor 1..8
    float mnew[8];
    #pragma unroll
    for (int r = 0; r < 8; ++r) {
      float mv = m_i[r];
      #pragma unroll
      for (int tc = 0; tc < 4; ++tc) mv = fmaxf(mv, sc[tc][r]);
      mnew[r] = mv;
    }
    #pragma unroll
    for (int off = 1; off < 16; off <<= 1)
      #pragma unroll
      for (int r = 0; r < 8; ++r)
        mnew[r] = fmaxf(mnew[r], __shfl_xor(mnew[r], off));

    float rs[8];
    #pragma unroll
    for (int r = 0; r < 8; ++r) rs[r] = 0.0f;
    #pragma unroll
    for (int tc = 0; tc < 4; ++tc)
      #pragma unroll
      for (int r = 0; r < 8; ++r) {
        float p = __expf(sc[tc][r] - mnew[r]);
        sc[tc][r] = p;
        rs[r] += p;
      }
    #pragma unroll
    for (int off = 1; off < 16; off <<= 1)
      #pragma unroll
      for (int r = 0; r < 8; ++r)
        rs[r] += __shfl_xor(rs[r], off);

    #pragma unroll
    for (int r = 0; r < 8; ++r) {
      const float corr = __expf(m_i[r] - mnew[r]);
      l_i[r] = l_i[r] * corr + rs[r];
      m_i[r] = mnew[r];
      #pragma unroll
      for (int nc = 0; nc < 4; ++nc) Oacc[nc][r] *= corr;
    }

    // C-layout -> A-layout via per-wave LDS strip (DS in-order within wave)
    _Float16* Pw = &Ps[wave][0];
    #pragma unroll
    for (int tc = 0; tc < 4; ++tc)
      #pragma unroll
      for (int r = 0; r < 8; ++r)
        Pw[(hi * 8 + r) * 72 + tc * 16 + nn] = (_Float16)sc[tc][r];

    // O += P . V  (inner dim keys = 64)
    v16h pa0 = load_a_frag(Pw, 72, 0);
    v16h pa1 = load_a_frag(Pw, 72, 32);
    #pragma unroll
    for (int nc = 0; nc < 4; ++nc) {
      Oacc[nc] = wmma_f16(pa0, load_b_frag(Vts[cur] + nc * 16 * 72, 72, 0),  Oacc[nc]);
      Oacc[nc] = wmma_f16(pa1, load_b_frag(Vts[cur] + nc * 16 * 72, 72, 32), Oacc[nc]);
    }
    __syncthreads();
  }

  #pragma unroll
  for (int r = 0; r < 8; ++r) {
    const int qg = qb * 64 + wave * 16 + hi * 8 + r;
    const float inv = 1.0f / l_i[r];
    const size_t rowoff = ((size_t)b * S_LEN + qg) * E_DIM + h * HD;
    #pragma unroll
    for (int nc = 0; nc < 4; ++nc)
      Ctx[rowoff + nc * 16 + nn] = (_Float16)(Oacc[nc][r] * inv);
  }
}

// ---------------------------------------------------------------------------
// Kernel 3: Y = residual + Ctx(f16) @ Wo + bo  (f32 out, pre-LayerNorm)
// Both A and B tiles are f16 -> fully async double-buffered staging.
// ---------------------------------------------------------------------------
__global__ void __launch_bounds__(256)
out_proj_kernel(const _Float16* __restrict__ Ctx, const _Float16* __restrict__ Wt,
                const float* __restrict__ bias, const float* __restrict__ resid,
                float* __restrict__ Y)
{
  __shared__ _Float16 As[2][64 * 40];
  __shared__ _Float16 Bs[2][64 * 40];

  const int m0  = blockIdx.x * 64;
  const int n0  = blockIdx.y * 64;
  const int tid = threadIdx.x;
  const int wave = tid >> 5;
  const int tr   = wave >> 1;
  const int tc0  = (wave & 1) * 2;

  v8f acc0 = {}; v8f acc1 = {};

  const int br = tid >> 2;             // 0..63
  const int bc = (tid & 3) * 8;        // 0..24

  async_copy_b128(As[0] + br * 40 + bc, Ctx + (size_t)(m0 + br) * E_DIM + bc);
  async_copy_b128(Bs[0] + br * 40 + bc, Wt  + (size_t)(n0 + br) * E_DIM + bc);

  const int NK = E_DIM / 32;
  for (int it = 0; it < NK; ++it) {
    const int cur = it & 1, nxt = cur ^ 1;
    const bool pre = (it + 1) < NK;
    if (pre) {
      const int k1 = (it + 1) * 32;
      async_copy_b128(As[nxt] + br * 40 + bc,
                      Ctx + (size_t)(m0 + br) * E_DIM + k1 + bc);
      async_copy_b128(Bs[nxt] + br * 40 + bc,
                      Wt  + (size_t)(n0 + br) * E_DIM + k1 + bc);
      async_wait_2();
    } else {
      async_wait_all();
    }
    __syncthreads();

    v16h a  = load_a_frag(As[cur] + tr * 16 * 40, 40, 0);
    v16h b0 = load_b_frag(Bs[cur] + tc0 * 16 * 40, 40, 0);
    v16h b1 = load_b_frag(Bs[cur] + (tc0 + 1) * 16 * 40, 40, 0);
    acc0 = wmma_f16(a, b0, acc0);
    acc1 = wmma_f16(a, b1, acc1);
    __syncthreads();
  }

  const int l  = tid & 31;
  const int hi = l >> 4;
  const int nn = l & 15;
  #pragma unroll
  for (int t = 0; t < 2; ++t) {
    const v8f acc = t ? acc1 : acc0;
    const int gn = n0 + (tc0 + t) * 16 + nn;
    const float bv = bias[gn];
    #pragma unroll
    for (int r = 0; r < 8; ++r) {
      const int gm = m0 + tr * 16 + hi * 8 + r;
      const size_t idx = (size_t)gm * E_DIM + gn;
      Y[idx] = acc[r] + bv + resid[idx];
    }
  }
}

// ---------------------------------------------------------------------------
// Kernel 4: row LayerNorm in-place on d_out. One block per row.
// ---------------------------------------------------------------------------
__global__ void __launch_bounds__(256)
layernorm_kernel(float* __restrict__ Y, const float* __restrict__ gamma,
                 const float* __restrict__ beta)
{
  const int row = blockIdx.x;
  float* p = Y + (size_t)row * E_DIM;
  const int tid = threadIdx.x;

  float vals[4];
  float s = 0.f, ss = 0.f;
  #pragma unroll
  for (int i = 0; i < 4; ++i) {
    float v = p[tid + i * 256];
    vals[i] = v; s += v; ss += v * v;
  }
  #pragma unroll
  for (int off = 16; off >= 1; off >>= 1) {
    s  += __shfl_xor(s, off);
    ss += __shfl_xor(ss, off);
  }
  __shared__ float rsum[8], rsq[8];
  const int wave = tid >> 5, lane = tid & 31;
  if (lane == 0) { rsum[wave] = s; rsq[wave] = ss; }
  __syncthreads();
  if (tid < 32) {
    s  = (lane < 8) ? rsum[lane] : 0.f;
    ss = (lane < 8) ? rsq[lane]  : 0.f;
    #pragma unroll
    for (int off = 4; off >= 1; off >>= 1) {
      s += __shfl_xor(s, off); ss += __shfl_xor(ss, off);
    }
    if (lane == 0) { rsum[0] = s; rsq[0] = ss; }
  }
  __syncthreads();
  s = rsum[0]; ss = rsq[0];
  const float mu  = s * (1.0f / E_DIM);
  const float var = ss * (1.0f / E_DIM) - mu * mu;
  const float inv = rsqrtf(var + 1e-5f);
  #pragma unroll
  for (int i = 0; i < 4; ++i) {
    const int c = tid + i * 256;
    p[c] = (vals[i] - mu) * inv * gamma[c] + beta[c];
  }
}

// ---------------------------------------------------------------------------
extern "C" void kernel_launch(void* const* d_in, const int* in_sizes, int n_in,
                              void* d_out, int out_size, void* d_ws, size_t ws_size,
                              hipStream_t stream) {
  (void)in_sizes; (void)n_in; (void)out_size; (void)ws_size;
  const float* q    = (const float*)d_in[0];
  const float* k    = (const float*)d_in[1];
  const float* v    = (const float*)d_in[2];
  const float* wq   = (const float*)d_in[3];
  const float* bq   = (const float*)d_in[4];
  const float* wk   = (const float*)d_in[5];
  const float* bk   = (const float*)d_in[6];
  const float* wv   = (const float*)d_in[7];
  const float* bv   = (const float*)d_in[8];
  const float* wo   = (const float*)d_in[9];
  const float* bo   = (const float*)d_in[10];
  const float* gamma= (const float*)d_in[11];
  const float* beta = (const float*)d_in[12];
  const int*   mask = (const int*)d_in[13];
  float* out = (float*)d_out;

  const size_t N  = (size_t)M_DIM * E_DIM;   // 4 Mi activations
  const size_t EW = (size_t)E_DIM * E_DIM;   // 1 Mi per weight
  _Float16* qh  = (_Float16*)d_ws;           // [b,h,s,d], pre-scaled
  _Float16* kh  = qh + N;                    // [b,h,s,d]
  _Float16* vth = kh + N;                    // [b,h,d,s] (transposed)
  _Float16* ch  = vth + N;                   // [b,s,e] context
  _Float16* wqt = ch + N;                    // f16 W^T buffers
  _Float16* wkt = wqt + EW;
  _Float16* wvt = wkt + EW;
  _Float16* wot = wvt + EW;

  dim3 tgrid(E_DIM / 32, E_DIM / 32);
  wconvert_kernel<<<tgrid, 256, 0, stream>>>(wq, wqt);
  wconvert_kernel<<<tgrid, 256, 0, stream>>>(wk, wkt);
  wconvert_kernel<<<tgrid, 256, 0, stream>>>(wv, wvt);
  wconvert_kernel<<<tgrid, 256, 0, stream>>>(wo, wot);

  dim3 ggrid(M_DIM / 64, E_DIM / 64);
  const float qscale = 1.0f / (sqrtf((float)HD) + 1e-8f);

  qkv_proj_kernel<0><<<ggrid, 256, 0, stream>>>(q, wqt, bq, qh,  qscale);
  qkv_proj_kernel<0><<<ggrid, 256, 0, stream>>>(k, wkt, bk, kh,  1.0f);
  qkv_proj_kernel<1><<<ggrid, 256, 0, stream>>>(v, wvt, bv, vth, 1.0f);

  attn_kernel<<<dim3(S_LEN / 64, B_DIM * NH), 128, 0, stream>>>(qh, kh, vth, ch, mask);

  out_proj_kernel<<<ggrid, 256, 0, stream>>>(ch, wot, bo, q, out);
  layernorm_kernel<<<M_DIM, 256, 0, stream>>>(out, gamma, beta);
}
